// AttentionWithoutEinops_11553462026637
// MI455X (gfx1250) — matrix operand order repaired and checked by reading, not compiled, not run on hardware
//
#include <hip/hip_runtime.h>

// ---------------------------------------------------------------------------
// Attention layer for MI455X (gfx1250, wave32, WMMA bf16 -> f32 accumulate).
// Pipeline: [prep: f32->bf16 converts, W_O transpose]
//           [qkv_proj: WMMA GEMM, B-tile staged in LDS via async DMA, V^T out]
//           [flash_attn: S^T trick -> lane-local P repack, online softmax]
//           [out_proj: WMMA GEMM, LDS-staged B -> fp32 d_out]
// ---------------------------------------------------------------------------

typedef __attribute__((ext_vector_type(16))) __bf16 v16bf;
typedef __attribute__((ext_vector_type(8)))  __bf16 v8bf;
typedef __attribute__((ext_vector_type(8)))  float  v8f;

#define B_  2
#define S_  2048
#define D_  1024
#define NH_ 16
#define DH_ 64

// LDS staging geometry: 64 B-rows x 32 bf16 (64B) padded to 80B row stride
// so half-wave lanes hit disjoint 4-bank windows on ds_load_b128.
#define BROW_   80
#define BUFSZ_  (64 * BROW_)   // 5120 B per buffer, x2 double buffer

static __device__ __forceinline__ v16bf join8(v8bf lo, v8bf hi) {
  return __builtin_shufflevector(lo, hi, 0,1,2,3,4,5,6,7,8,9,10,11,12,13,14,15);
}
static __device__ __forceinline__ v16bf load16(const __bf16* p) { return *(const v16bf*)p; }
static __device__ __forceinline__ v8bf  load8 (const __bf16* p) { return *(const v8bf*)p; }

static __device__ __forceinline__ v8f wmma_bf16(v16bf a, v16bf b, v8f c) {
  // D = A(16x32 bf16) * B(32x16 bf16) + C(16x16 f32)
  return __builtin_amdgcn_wmma_f32_16x16x32_bf16(false, a, false, b, (short)0, c, false, false);
}

// Async Global->LDS b128 copy (ASYNCcnt-tracked DMA path, CDNA5).
static __device__ __forceinline__ void async_g2l_b128(unsigned lds_off,
                                                      const void* gptr) {
  unsigned long long ga = (unsigned long long)(size_t)gptr;
  asm volatile("global_load_async_to_lds_b128 %0, %1, off"
               :: "v"(lds_off), "v"(ga) : "memory");
}
static __device__ __forceinline__ void wait_async0() {
  asm volatile("s_wait_asynccnt 0" ::: "memory");
}

// Stage one 64x32 bf16 B-tile (rows n0..n0+63, k=k0..k0+31) into LDS buffer.
// 256 threads x one b128 async load each.
static __device__ __forceinline__ void stage_btile(const __bf16* W, int n0,
                                                   int k0, char* smem, int buf,
                                                   int tid) {
  const int row   = tid >> 2;   // 0..63
  const int chunk = tid & 3;    // 4 x 16B per row
  const __bf16* g = W + (size_t)(n0 + row) * D_ + k0 + chunk * 8;
  unsigned lds = (unsigned)(size_t)(smem + buf * BUFSZ_ + row * BROW_ + chunk * 16);
  async_g2l_b128(lds, g);
}

// Read B operand (one 16-col subtile) for this lane from the staged buffer.
static __device__ __forceinline__ v16bf ldsd_b(const char* bb, int row) {
  const char* p = bb + row * BROW_;
  return join8(*(const v8bf*)p, *(const v8bf*)(p + 16));
}

// ---------------------------- prep kernels ---------------------------------

__global__ void cvt_f32_bf16_kernel(const float* __restrict__ src,
                                    __bf16* __restrict__ dst, int n) {
  int i = blockIdx.x * blockDim.x + threadIdx.x;
  if (i < n) dst[i] = (__bf16)src[i];
}

// W_O is [k=(head*64+h)][d] row-major (1024x1024); produce bf16 [d][k].
__global__ void transpose_cvt_kernel(const float* __restrict__ src,
                                     __bf16* __restrict__ dst) {
  int i = blockIdx.x * blockDim.x + threadIdx.x;   // i over 1024*1024
  int k = i >> 10;
  int d = i & 1023;
  dst[(size_t)d * 1024 + k] = (__bf16)src[i];
}

// ---------------------------- QKV projection --------------------------------
// out[m, n] = sum_k resb[m,k] * W[n,k]   (W row-major [n=(head*64+h)][k=d])
// grid: (N/64, M/128, 3); 8 waves/block; B tile shared block-wide via LDS DMA.
// Q,K stored [b][head][p][h]; V stored transposed [b][head][h][p].

__global__ void __launch_bounds__(256)
qkv_proj_kernel(const __bf16* __restrict__ resb,
                const __bf16* __restrict__ WQb, const __bf16* __restrict__ WKb,
                const __bf16* __restrict__ WVb,
                __bf16* __restrict__ Qws, __bf16* __restrict__ Kws,
                __bf16* __restrict__ Vtws) {
  __shared__ __align__(16) char smem[2 * BUFSZ_];
  const int which = blockIdx.z;  // 0=Q 1=K 2=V
  const __bf16* W = which == 0 ? WQb : (which == 1 ? WKb : WVb);
  const int tid  = threadIdx.x;
  const int lane = tid & 31;
  const int wave = tid >> 5;
  const int lh   = lane >> 4;       // half-wave: 0 or 1
  const int lm   = lane & 15;
  const int n0 = blockIdx.x * 64;                 // one head (64 cols) per block.x
  const int m0 = blockIdx.y * 128 + wave * 16;

  v8f acc0 = {}, acc1 = {}, acc2 = {}, acc3 = {};
  const __bf16* arow = resb + (size_t)(m0 + lm) * D_;

  stage_btile(W, n0, 0, smem, 0, tid);
  for (int k0 = 0; k0 < D_; k0 += 32) {
    const int buf = (k0 >> 5) & 1;
    wait_async0();          // our own DMA done
    __syncthreads();        // everyone's DMA done; prior buf reads all retired
    if (k0 + 32 < D_) stage_btile(W, n0, k0 + 32, smem, buf ^ 1, tid);

    // A operand: lane row m0+lm; halves own K 0..15 / 16..31
    v16bf a = load16(arow + k0 + lh * 16);
    // B operand from LDS: lane col n0+sub*16+lm, halves own K 0..15 / 16..31
    const char* bb = smem + buf * BUFSZ_ + lh * 32;
    v16bf b0 = ldsd_b(bb,  0 + lm);
    v16bf b1 = ldsd_b(bb, 16 + lm);
    v16bf b2 = ldsd_b(bb, 32 + lm);
    v16bf b3 = ldsd_b(bb, 48 + lm);
    acc0 = wmma_bf16(a, b0, acc0);
    acc1 = wmma_bf16(a, b1, acc1);
    acc2 = wmma_bf16(a, b2, acc2);
    acc3 = wmma_bf16(a, b3, acc3);
  }
  const int head = n0 >> 6;
  const int b    = m0 / S_;
  const int p0   = m0 % S_;
  #pragma unroll
  for (int nt = 0; nt < 4; ++nt) {
    v8f acc = nt == 0 ? acc0 : nt == 1 ? acc1 : nt == 2 ? acc2 : acc3;
    const int h = nt * 16 + lm;
    #pragma unroll
    for (int r = 0; r < 8; ++r) {
      const int p = p0 + r + lh * 8;    // D layout: vgpr r, half -> row
      __bf16 val = (__bf16)acc[r];
      if (which == 2) {
        Vtws[((size_t)(b * NH_ + head) * DH_ + h) * S_ + p] = val;
      } else {
        __bf16* O = which == 0 ? Qws : Kws;
        O[((size_t)(b * NH_ + head) * S_ + p) * DH_ + h] = val;
      }
    }
  }
}

// ---------------------------- flash attention -------------------------------
// Per wave: 16 q rows. Scores computed transposed: S^T = K_tile x Q^T so the
// softmaxed D-matrix is already in P*V A-operand lane layout (lane = q col).
// grid: (S/128, NH, B); 8 waves/block.

__global__ void __launch_bounds__(256)
flash_attn_kernel(const __bf16* __restrict__ Qws, const __bf16* __restrict__ Kws,
                  const __bf16* __restrict__ Vtws, __bf16* __restrict__ attn) {
  const int lane = threadIdx.x & 31;
  const int wave = threadIdx.x >> 5;
  const int lh   = lane >> 4;
  const int lm   = lane & 15;
  const int b    = blockIdx.z;
  const int head = blockIdx.y;
  const int q0   = blockIdx.x * 128 + wave * 16;
  const int q    = q0 + lm;                     // this lane's q column

  const __bf16* Qbase = Qws  + (size_t)(b * NH_ + head) * S_  * DH_;
  const __bf16* Kbase = Kws  + (size_t)(b * NH_ + head) * S_  * DH_;
  const __bf16* Vbase = Vtws + (size_t)(b * NH_ + head) * DH_ * S_;

  // Q as B operand (K-dim = dh), two dh steps of 32
  const __bf16* qrow = Qbase + (size_t)q * DH_;
  v16bf bq0 = join8(load8(qrow + 0  + lh * 8), load8(qrow + 16 + lh * 8));
  v16bf bq1 = join8(load8(qrow + 32 + lh * 8), load8(qrow + 48 + lh * 8));

  v8f o0 = {}, o1 = {}, o2 = {}, o3 = {};
  float m_run = -1e30f, l_run = 0.0f;
  const float scale = 0.125f;   // 1/sqrt(64)

  const int kend = q0 + 16;     // causal: keys <= q0+15
  for (int k0 = 0; k0 < kend; k0 += 32) {
    if (k0 + 32 < kend)
      __builtin_prefetch(Kbase + (size_t)(k0 + 32 + lm) * DH_, 0, 3);

    // Two 16-key score subtiles: S^T tile = K_rows x Q^T
    v8f st0 = {}, st1 = {};
    {
      const __bf16* krow = Kbase + (size_t)(k0 + lm) * DH_;
      v16bf ka0 = join8(load8(krow + lh * 8),      load8(krow + 16 + lh * 8));
      v16bf ka1 = join8(load8(krow + 32 + lh * 8), load8(krow + 48 + lh * 8));
      st0 = wmma_bf16(ka0, bq0, st0);
      st0 = wmma_bf16(ka1, bq1, st0);
    }
    {
      const __bf16* krow = Kbase + (size_t)(k0 + 16 + lm) * DH_;
      v16bf ka0 = join8(load8(krow + lh * 8),      load8(krow + 16 + lh * 8));
      v16bf ka1 = join8(load8(krow + 32 + lh * 8), load8(krow + 48 + lh * 8));
      st1 = wmma_bf16(ka0, bq0, st1);
      st1 = wmma_bf16(ka1, bq1, st1);
    }

    // Per-lane 16 scores for column q: lane owns k offsets
    //   half 0: {0..7, 16..23}, half 1: {8..15, 24..31}  (== PV A-layout)
    float sv[16];
    float mx = -1e30f;
    #pragma unroll
    for (int r = 0; r < 8; ++r) {
      const int kk0 = k0 + r + lh * 8;
      const int kk1 = k0 + 16 + r + lh * 8;
      sv[r]     = (kk0 <= q) ? st0[r] * scale : -1e30f;
      sv[8 + r] = (kk1 <= q) ? st1[r] * scale : -1e30f;
      mx = fmaxf(mx, fmaxf(sv[r], sv[8 + r]));
    }
    mx = fmaxf(mx, __shfl_xor(mx, 16, 32));     // join half-wave partner
    const float m_new = fmaxf(m_run, mx);
    float sum = 0.0f;
    #pragma unroll
    for (int i = 0; i < 16; ++i) { sv[i] = __expf(sv[i] - m_new); sum += sv[i]; }
    sum += __shfl_xor(sum, 16, 32);
    const float alpha = __expf(m_run - m_new);
    l_run = l_run * alpha + sum;
    m_run = m_new;

    // Rescale O accumulators: alpha per q-row; O rows live across vgprs
    float alr[8];
    #pragma unroll
    for (int r = 0; r < 8; ++r) alr[r] = __shfl(alpha, r + lh * 8, 32);
    #pragma unroll
    for (int r = 0; r < 8; ++r) {
      o0[r] *= alr[r]; o1[r] *= alr[r]; o2[r] *= alr[r]; o3[r] *= alr[r];
    }

    // Pack P (lane-local!) into PV A operand
    v16bf ap;
    #pragma unroll
    for (int i = 0; i < 16; ++i) ap[i] = (__bf16)sv[i];

    // V^T rows are contiguous in key dim -> B operand direct loads
    const int kb = k0 + lh * 8;
    v16bf bv0 = join8(load8(Vbase + (size_t)( 0 + lm) * S_ + kb), load8(Vbase + (size_t)( 0 + lm) * S_ + kb + 16));
    v16bf bv1 = join8(load8(Vbase + (size_t)(16 + lm) * S_ + kb), load8(Vbase + (size_t)(16 + lm) * S_ + kb + 16));
    v16bf bv2 = join8(load8(Vbase + (size_t)(32 + lm) * S_ + kb), load8(Vbase + (size_t)(32 + lm) * S_ + kb + 16));
    v16bf bv3 = join8(load8(Vbase + (size_t)(48 + lm) * S_ + kb), load8(Vbase + (size_t)(48 + lm) * S_ + kb + 16));
    o0 = wmma_bf16(ap, bv0, o0);
    o1 = wmma_bf16(ap, bv1, o1);
    o2 = wmma_bf16(ap, bv2, o2);
    o3 = wmma_bf16(ap, bv3, o3);
  }

  // Normalize and store attn_out row-major [(b,p)][(head*64+h)] as bf16
  const float inv = 1.0f / l_run;
  float invr[8];
  #pragma unroll
  for (int r = 0; r < 8; ++r) invr[r] = __shfl(inv, r + lh * 8, 32);
  #pragma unroll
  for (int nt = 0; nt < 4; ++nt) {
    v8f o = nt == 0 ? o0 : nt == 1 ? o1 : nt == 2 ? o2 : o3;
    #pragma unroll
    for (int r = 0; r < 8; ++r) {
      const int p = q0 + r + lh * 8;
      attn[(size_t)(b * S_ + p) * D_ + head * DH_ + nt * 16 + lm] =
          (__bf16)(o[r] * invr[r]);
    }
  }
}

// ---------------------------- output projection -----------------------------
// out[m, d] = sum_k attn[m,k] * WOt[d][k]   (WOt pre-transposed bf16 [d][k])
// grid: (D/64, M/128); LDS-staged B; fp32 stores straight to d_out.

__global__ void __launch_bounds__(256)
out_proj_kernel(const __bf16* __restrict__ attn, const __bf16* __restrict__ WOt,
                float* __restrict__ out) {
  __shared__ __align__(16) char smem[2 * BUFSZ_];
  const int tid  = threadIdx.x;
  const int lane = tid & 31;
  const int wave = tid >> 5;
  const int lh   = lane >> 4;
  const int lm   = lane & 15;
  const int n0 = blockIdx.x * 64;
  const int m0 = blockIdx.y * 128 + wave * 16;

  v8f acc0 = {}, acc1 = {}, acc2 = {}, acc3 = {};
  const __bf16* arow = attn + (size_t)(m0 + lm) * D_;

  stage_btile(WOt, n0, 0, smem, 0, tid);
  for (int k0 = 0; k0 < D_; k0 += 32) {
    const int buf = (k0 >> 5) & 1;
    wait_async0();
    __syncthreads();
    if (k0 + 32 < D_) stage_btile(WOt, n0, k0 + 32, smem, buf ^ 1, tid);

    v16bf a = load16(arow + k0 + lh * 16);
    const char* bb = smem + buf * BUFSZ_ + lh * 32;
    v16bf b0 = ldsd_b(bb,  0 + lm);
    v16bf b1 = ldsd_b(bb, 16 + lm);
    v16bf b2 = ldsd_b(bb, 32 + lm);
    v16bf b3 = ldsd_b(bb, 48 + lm);
    acc0 = wmma_bf16(a, b0, acc0);
    acc1 = wmma_bf16(a, b1, acc1);
    acc2 = wmma_bf16(a, b2, acc2);
    acc3 = wmma_bf16(a, b3, acc3);
  }
  #pragma unroll
  for (int nt = 0; nt < 4; ++nt) {
    v8f acc = nt == 0 ? acc0 : nt == 1 ? acc1 : nt == 2 ? acc2 : acc3;
    const int d = n0 + nt * 16 + lm;
    #pragma unroll
    for (int r = 0; r < 8; ++r) {
      const int m = m0 + r + lh * 8;
      out[(size_t)m * D_ + d] = acc[r];
    }
  }
}

// ---------------------------- host launch -----------------------------------

extern "C" void kernel_launch(void* const* d_in, const int* in_sizes, int n_in,
                              void* d_out, int out_size, void* d_ws, size_t ws_size,
                              hipStream_t stream) {
  const float* residual = (const float*)d_in[0];
  const float* W_Q = (const float*)d_in[1];
  const float* W_K = (const float*)d_in[2];
  const float* W_V = (const float*)d_in[3];
  const float* W_O = (const float*)d_in[4];
  (void)in_sizes; (void)n_in; (void)out_size; (void)ws_size;

  char* ws = (char*)d_ws;
  const size_t MB = 1024ull * 1024ull;
  __bf16* resb = (__bf16*)(ws +  0 * MB);   //  8 MiB : residual bf16 [4096][1024]
  __bf16* WQb  = (__bf16*)(ws +  8 * MB);   //  2 MiB : W_Q bf16 [(n h)][d]
  __bf16* WKb  = (__bf16*)(ws + 10 * MB);   //  2 MiB
  __bf16* WVb  = (__bf16*)(ws + 12 * MB);   //  2 MiB
  __bf16* WOtb = (__bf16*)(ws + 14 * MB);   //  2 MiB : W_O^T bf16 [d][(n h)]
  __bf16* Qws  = (__bf16*)(ws + 16 * MB);   //  8 MiB : [b][h][p][dh]
  __bf16* Kws  = (__bf16*)(ws + 24 * MB);   //  8 MiB : [b][h][p][dh]
  __bf16* Vtws = (__bf16*)(ws + 32 * MB);   //  8 MiB : [b][h][dh][p]
  __bf16* attn = (__bf16*)(ws + 40 * MB);   //  8 MiB : [(b p)][(h dh)]

  const int nres = B_ * S_ * D_;            // 4,194,304
  const int nw   = NH_ * DH_ * D_;          // 1,048,576
  cvt_f32_bf16_kernel<<<nres / 256, 256, 0, stream>>>(residual, resb, nres);
  cvt_f32_bf16_kernel<<<nw   / 256, 256, 0, stream>>>(W_Q, WQb, nw);
  cvt_f32_bf16_kernel<<<nw   / 256, 256, 0, stream>>>(W_K, WKb, nw);
  cvt_f32_bf16_kernel<<<nw   / 256, 256, 0, stream>>>(W_V, WVb, nw);
  transpose_cvt_kernel<<<nw  / 256, 256, 0, stream>>>(W_O, WOtb);

  dim3 g1(D_ / 64, (B_ * S_) / 128, 3);
  qkv_proj_kernel<<<g1, 256, 0, stream>>>(resb, WQb, WKb, WVb, Qws, Kws, Vtws);

  dim3 g2(S_ / 128, NH_, B_);
  flash_attn_kernel<<<g2, 256, 0, stream>>>(Qws, Kws, Vtws, attn);

  dim3 g3(D_ / 64, (B_ * S_) / 128);
  out_proj_kernel<<<g3, 256, 0, stream>>>(attn, WOtb, (float*)d_out);
}
